// EnhancedDRKANTreeNet_15805479649633
// MI455X (gfx1250) — matrix operand order repaired and checked by reading, not compiled, not run on hardware
//
#include <hip/hip_runtime.h>
#include <hip/hip_bf16.h>

typedef __attribute__((ext_vector_type(16))) __bf16 v16bf;
typedef __attribute__((ext_vector_type(8)))  float  v8f;

#define D_MODEL   1024
#define N_TOK     8192
#define KSTEPS    128                         // augmented K = 4096 = 128 * 32
#define BSW_BYTES (4096ull * 1024ull * 2ull)  // 8 MB
#define ASW_BYTES (8192ull * 4096ull * 2ull)  // 64 MB

__device__ __forceinline__ unsigned short f2bf(float f) {
    unsigned int u = __float_as_uint(f);
    unsigned int r = u + 0x7FFFu + ((u >> 16) & 1u);   // round-to-nearest-even
    return (unsigned short)(r >> 16);
}

__device__ __forceinline__ float basis_val(float xv, int f) {
    // f == 0 -> x itself; f == 1..3 -> normalized quadratic basis at grid {-1,0,1}
    if (f == 0) return xv;
    float d0 = fabsf(xv + 1.f), d1 = fabsf(xv), d2 = fabsf(xv - 1.f);
    float b0 = (d0 < 1.f) ? (1.f - d0) * (1.f - d0) : 0.f;
    float b1 = (d1 < 1.f) ? (1.f - d1) * (1.f - d1) : 0.f;
    float b2 = (d2 < 1.f) ? (1.f - d2) * (1.f - d2) : 0.f;
    float rs = 1.f / (b0 + b1 + b2 + 1e-6f);
    float b  = (f == 1) ? b0 : ((f == 2) ? b1 : b2);
    return b * rs;
}

// ---------------------------------------------------------------------------
// Kernel 1: swizzled bf16 augmented-B  [kt][n][half][16] (lane-contiguous 32B).
//   ka = kt*32 + half*16 + q ; f = ka&3 ; k0 = ka>>2
// ---------------------------------------------------------------------------
__global__ void prep_b_kernel(const float* __restrict__ bw,
                              const float* __restrict__ sw,
                              unsigned short* __restrict__ bsw) {
    int idx  = blockIdx.x * 256 + threadIdx.x;      // 0 .. 4096*1024-1
    int q    = idx & 15;
    int half = (idx >> 4) & 1;
    int n    = (idx >> 5) & 1023;
    int kt   = idx >> 15;
    int ka   = kt * 32 + half * 16 + q;
    int k0   = ka >> 2;
    int f    = ka & 3;
    float v  = (f == 0) ? bw[(long)n * 1024 + k0]
                        : sw[((long)n * 1024 + k0) * 3 + (f - 1)];
    bsw[idx] = f2bf(v);
}

// ---------------------------------------------------------------------------
// Kernel 2: swizzled bf16 augmented-A panels in WMMA A-fragment order.
//   element index e: q = e&15, lane = (e>>4)&31, kt = (e>>9)&127, mt = e>>16
//   row = mt*16 + (lane&15); augK = half*16 + q (matches B layout)
// ---------------------------------------------------------------------------
__global__ void prep_a_kernel(const float* __restrict__ x,
                              unsigned short* __restrict__ asw) {
    long e   = (long)blockIdx.x * 256 + threadIdx.x;   // 0 .. 8192*4096-1
    int q    = (int)(e & 15);
    int lane = (int)((e >> 4) & 31);
    int kt   = (int)((e >> 9) & 127);
    long mt  = e >> 16;
    int half = lane >> 4;
    int row  = (int)mt * 16 + (lane & 15);
    int aug  = half * 16 + q;
    int ka   = kt * 32 + aug;
    int k0   = ka >> 2;
    int f    = ka & 3;
    float xv = x[(long)row * D_MODEL + k0];
    asw[e]   = f2bf(basis_val(xv, f));
}

// ---------------------------------------------------------------------------
// Kernel 3: GEMM, no LDS / no barriers. 256 thr = 8 waves.
// Block: M-tile 32 (2 subtiles), wave w owns N in [w*128, w*128+128).
// Each B fragment is reused by 2 WMMAs (both M-subtiles).
// ---------------------------------------------------------------------------
__global__ void __launch_bounds__(256)
kan_gemm_kernel(const unsigned short* __restrict__ asw,
                const unsigned short* __restrict__ bsw,
                float* __restrict__ pre) {
    const int tid  = threadIdx.x;
    const int wave = tid >> 5;
    const int lane = tid & 31;
    const int col  = lane & 15;
    const int half = lane >> 4;
    const long mt0 = (long)blockIdx.x * 2;

    // A: per-kt stride 1024 B; second M-subtile at +131072 B (immediate offset)
    const char* ap = (const char*)asw + (mt0 * 128 * 32 + lane) * 32;
    // B: per-kt stride 65536 B; n-tile t at +t*1024 B (immediate offset)
    const char* bp = (const char*)bsw +
                     (long)(((wave * 128 + col) * 2 + half) * 32);

    v8f acc[16];
#pragma unroll
    for (int t = 0; t < 16; ++t)
        acc[t] = (v8f){0.f,0.f,0.f,0.f,0.f,0.f,0.f,0.f};

    for (int kt = 0; kt < KSTEPS; ++kt) {
        union { uint4 q[2]; v16bf v; } a0, a1;
        a0.q[0] = *(const uint4*)(ap);
        a0.q[1] = *(const uint4*)(ap + 16);
        a1.q[0] = *(const uint4*)(ap + 131072);
        a1.q[1] = *(const uint4*)(ap + 131072 + 16);
        __builtin_prefetch(ap + 1024, 0, 3);
#pragma unroll
        for (int t = 0; t < 8; ++t) {
            union { uint4 q[2]; v16bf v; } b;
            b.q[0] = *(const uint4*)(bp + t * 1024);
            b.q[1] = *(const uint4*)(bp + t * 1024 + 16);
            acc[t]     = __builtin_amdgcn_wmma_f32_16x16x32_bf16(
                             false, a0.v, false, b.v, (short)0, acc[t],
                             false, false);
            acc[8 + t] = __builtin_amdgcn_wmma_f32_16x16x32_bf16(
                             false, a1.v, false, b.v, (short)0, acc[8 + t],
                             false, false);
        }
        __builtin_prefetch(bp + 65536, 0, 3);
        ap += 1024;
        bp += 65536;
    }

    // store pre-LN activations (C layout: lane<16 -> M=v, lane>=16 -> M=v+8)
    const long row0 = mt0 * 16;
#pragma unroll
    for (int t = 0; t < 8; ++t) {
        int n = wave * 128 + t * 16 + col;
#pragma unroll
        for (int v = 0; v < 8; ++v) {
            int r = v + half * 8;
            pre[(row0 + r) * D_MODEL + n]      = acc[t][v];
            pre[(row0 + 16 + r) * D_MODEL + n] = acc[8 + t][v];
        }
    }
}

// ---------------------------------------------------------------------------
// Kernel 4: LayerNorm + SE gating. 8 rows per block, 256 threads (8 waves).
// ---------------------------------------------------------------------------
__global__ void __launch_bounds__(256)
kan_epilogue_kernel(const float* __restrict__ pre,
                    const float* __restrict__ lnw, const float* __restrict__ lnb,
                    const float* __restrict__ w1,  const float* __restrict__ b1,
                    const float* __restrict__ w2,  const float* __restrict__ b2,
                    float* __restrict__ out) {
    __shared__ float srow[8 * D_MODEL];   // 32 KB
    __shared__ float sh[8 * 32];

    const int tid   = threadIdx.x;
    const long base = (long)blockIdx.x * 8 * D_MODEL;

    for (int e = tid; e < 8 * D_MODEL; e += 256) srow[e] = pre[base + e];
    __syncthreads();

    // LayerNorm: one wave (32 lanes) per row
    const int row = tid >> 5, lane = tid & 31;
    float s = 0.f, ss = 0.f;
    for (int i = lane; i < D_MODEL; i += 32) {
        float v = srow[row * D_MODEL + i];
        s += v; ss += v * v;
    }
#pragma unroll
    for (int m = 16; m; m >>= 1) {
        s  += __shfl_xor(s,  m, 32);
        ss += __shfl_xor(ss, m, 32);
    }
    float mu  = s * (1.f / D_MODEL);
    float var = ss * (1.f / D_MODEL) - mu * mu;
    float inv = rsqrtf(var + 1e-5f);
    for (int i = lane; i < D_MODEL; i += 32) {
        float v = (srow[row * D_MODEL + i] - mu) * inv * lnw[i] + lnb[i];
        srow[row * D_MODEL + i] = v;
    }
    __syncthreads();

    // SE hidden: 8 rows x 32 hidden = 256 tasks, one per thread
    {
        int r = tid >> 5, j = tid & 31;
        float a = b1[j];
        const float* wr = w1 + (long)j * D_MODEL;
        for (int i = 0; i < D_MODEL; ++i) a += srow[r * D_MODEL + i] * wr[i];
        sh[r * 32 + j] = fmaxf(a, 0.f);
    }
    __syncthreads();

    // gate + store
    for (int e = tid; e < 8 * D_MODEL; e += 256) {
        int r = e >> 10, c = e & 1023;
        float z = b2[c];
        const float* wc = w2 + (long)c * 32;
#pragma unroll
        for (int j = 0; j < 32; ++j) z += sh[r * 32 + j] * wc[j];
        float g = 1.f / (1.f + __expf(-z));
        out[base + e] = srow[e] * g;
    }
}

// ---------------------------------------------------------------------------
extern "C" void kernel_launch(void* const* d_in, const int* in_sizes, int n_in,
                              void* d_out, int out_size, void* d_ws, size_t ws_size,
                              hipStream_t stream) {
    (void)in_sizes; (void)n_in; (void)out_size; (void)ws_size;
    const float* x   = (const float*)d_in[0];
    const float* bw  = (const float*)d_in[1];
    const float* sw  = (const float*)d_in[2];
    const float* lnw = (const float*)d_in[3];
    const float* lnb = (const float*)d_in[4];
    const float* w1  = (const float*)d_in[5];
    const float* b1  = (const float*)d_in[6];
    const float* w2  = (const float*)d_in[7];
    const float* b2  = (const float*)d_in[8];

    unsigned short* bsw = (unsigned short*)d_ws;
    unsigned short* asw = (unsigned short*)((char*)d_ws + BSW_BYTES);
    float* pre = (float*)((char*)d_ws + BSW_BYTES + ASW_BYTES);
    float* out = (float*)d_out;

    // 1) weights -> swizzled bf16 augmented B (4096 x 1024)
    prep_b_kernel<<<(4096 * 1024) / 256, 256, 0, stream>>>(bw, sw, bsw);
    // 2) x -> basis -> swizzled bf16 augmented A (8192 x 4096)
    prep_a_kernel<<<(int)(8192ll * 4096 / 256), 256, 0, stream>>>(x, asw);
    // 3) fused base+spline GEMM (8192 x 1024 x K=4096), barrier-free bf16 WMMA
    kan_gemm_kernel<<<N_TOK / 32, 256, 0, stream>>>(asw, bsw, pre);
    // 4) LayerNorm + SE gating
    kan_epilogue_kernel<<<N_TOK / 8, 256, 0, stream>>>(pre, lnw, lnb,
                                                       w1, b1, w2, b2, out);
}
